// MultiHeadAttention_35407710388794
// MI455X (gfx1250) — matrix operand order repaired and checked by reading, not compiled, not run on hardware
//
#include <hip/hip_runtime.h>

// ---------------- types ----------------
typedef __attribute__((ext_vector_type(16))) __bf16 v16bf;
typedef __attribute__((ext_vector_type(8)))  __bf16 v8bf;
typedef __attribute__((ext_vector_type(8)))  float  v8f;

union V16 { v16bf v; v8bf h[2]; };

__device__ __forceinline__ __bf16 f2bf(float f) { return (__bf16)f; }

__device__ __forceinline__ V16 loadA(const __bf16* rowp, int kc) {
    // A-operand (16x32 bf16): elements 0..7 = K+0..7 (lo lanes) / K+8..15 (hi lanes),
    // elements 8..15 = +16 from that. rowp is pre-offset by (hi?8:0).
    V16 a;
    a.h[0] = *(const v8bf*)(rowp + kc * 32);
    a.h[1] = *(const v8bf*)(rowp + kc * 32 + 16);
    return a;
}
__device__ __forceinline__ V16 loadB(const __bf16* bp) {
    // B-operand (32x16 bf16): elements 0..7 = K+0..7 (lo) / K+8..15 (hi), elements 8..15 = +16 (pre-offset by hi?8:0).
    V16 b;
    b.h[0] = *(const v8bf*)(bp);
    b.h[1] = *(const v8bf*)(bp + 16);
    return b;
}
#define WMMA_BF16(a, b, c) \
    __builtin_amdgcn_wmma_f32_16x16x32_bf16(false, (a), false, (b), (short)0, (c), false, false)

#define Bn 8
#define Ln 1024
#define Cn 768
#define Hn 6
#define Dn 128

// ---------------- elementwise f32 -> bf16 ----------------
__global__ void cvt_bf16(const float* __restrict__ src, unsigned short* __restrict__ dst_u, int n) {
    __bf16* dst = (__bf16*)dst_u;
    int i = blockIdx.x * 256 + threadIdx.x;
    if (i < n) dst[i] = f2bf(src[i]);
}

// ---------------- a1 = (adj>0) | eye, as bitmask rows (B,L,32 dwords) ----------------
__global__ void build_a1(const int* __restrict__ adj, unsigned int* __restrict__ a1) {
    int t = blockIdx.x * 256 + threadIdx.x;      // B*L*32 threads
    int word = t & 31;
    int l = (t >> 5) & (Ln - 1);
    int b = t >> 15;
    const int4* p = (const int4*)(adj + ((size_t)(b * Ln + l)) * Ln + word * 32);
    unsigned w = 0;
#pragma unroll
    for (int j = 0; j < 8; ++j) {
        int4 v = p[j];
        if (v.x > 0) w |= 1u << (4 * j + 0);
        if (v.y > 0) w |= 1u << (4 * j + 1);
        if (v.z > 0) w |= 1u << (4 * j + 2);
        if (v.w > 0) w |= 1u << (4 * j + 3);
    }
    int sc = l - word * 32;
    if (sc >= 0 && sc < 32) w |= (1u << sc);     // self loop
    a1[t] = w;
}

// ---------------- a2 bitset closure: one wave per (b,l) row ----------------
__global__ void build_a2(const unsigned int* __restrict__ a1, unsigned int* __restrict__ a2) {
    int lane = threadIdx.x & 31;
    int wg = blockIdx.x * 4 + (threadIdx.x >> 5);    // B*L waves
    int l = wg & (Ln - 1);
    int b = wg >> 10;
    const unsigned int* base = a1 + (size_t)b * Ln * 32;
    unsigned sel = base[l * 32 + lane];
    unsigned acc = 0;
    for (int m = 0; m < Ln; ++m) {
        unsigned wsel = (unsigned)__shfl((int)sel, m >> 5, 32);
        int bit = (int)((wsel >> (m & 31)) & 1u);
        if (__builtin_amdgcn_readfirstlane(bit))     // uniform scalar branch
            acc |= base[m * 32 + lane];
    }
    a2[(size_t)wg * 32 + lane] = acc;
}

// ---------------- pack 6 head-mask bits per (b,l,m) byte ----------------
__global__ void build_mask(const int* __restrict__ distance,
                           const unsigned int* __restrict__ a1,
                           const unsigned int* __restrict__ a2,
                           unsigned char* __restrict__ maskb) {
    size_t t = (size_t)blockIdx.x * 256 + threadIdx.x;   // B*L*L
    int m = (int)(t & (Ln - 1));
    int l = (int)((t >> 10) & (Ln - 1));
    int b = (int)(t >> 20);
    int d = distance[t];
    size_t bl = (size_t)(b * Ln + l);
    unsigned a1b = (a1[bl * 32 + (m >> 5)] >> (m & 31)) & 1u;
    unsigned a2b = (a2[bl * 32 + (m >> 5)] >> (m & 31)) & 1u;
    unsigned by = (unsigned)(l == m) | (a1b << 1) | (a2b << 2)
                | ((d <= 2) ? 8u : 0u) | ((d <= 3) ? 16u : 0u) | 32u;
    maskb[t] = (unsigned char)by;
}

// ---------------- QKV GEMM: 32x64 tile per wave, scatter to Q,K,Vt bf16 ----------------
__global__ void __launch_bounds__(128, 1)
gemm_qkv(const unsigned short* __restrict__ xb_u,
         const unsigned short* __restrict__ wb_u,
         unsigned short* __restrict__ Qg_u,
         unsigned short* __restrict__ Kg_u,
         unsigned short* __restrict__ Vt_u) {
    const __bf16* xb = (const __bf16*)xb_u;
    const __bf16* wb = (const __bf16*)wb_u;
    __bf16* Qg = (__bf16*)Qg_u; __bf16* Kg = (__bf16*)Kg_u; __bf16* Vt = (__bf16*)Vt_u;

    int lane = threadIdx.x & 31;
    int wg = blockIdx.x * 4 + (threadIdx.x >> 5);   // 256*36 waves
    int mt2 = wg & 255;                             // 32-row block
    int ng = wg >> 8;                               // 0..35, 64 cols each
    int ln = lane & 15;
    bool hi = lane >= 16;

    v8f c0[4], c1[4];
#pragma unroll
    for (int j = 0; j < 4; ++j)
#pragma unroll
        for (int e = 0; e < 8; ++e) { c0[j][e] = 0.f; c1[j][e] = 0.f; }

    const __bf16* ap0 = xb + (size_t)(mt2 * 32 + ln) * Cn + (hi ? 8 : 0);
    const __bf16* ap1 = ap0 + (size_t)16 * Cn;
    const __bf16* bb0 = wb + (size_t)(ng * 64 + ln) * Cn + (hi ? 8 : 0);

    V16 a0 = loadA(ap0, 0), a1 = loadA(ap1, 0);
#pragma unroll 2
    for (int kc = 0; kc < Cn / 32; ++kc) {
        V16 bf[4];
#pragma unroll
        for (int j = 0; j < 4; ++j)
            bf[j] = loadB(bb0 + (size_t)(j * 16) * Cn + kc * 32);
        // single-path double buffering: on the last iteration wrap to kc=0,
        // loaded value is discarded (address stays in-bounds).
        int kn = (kc == Cn / 32 - 1) ? 0 : kc + 1;
        V16 a0n = loadA(ap0, kn), a1n = loadA(ap1, kn);
#pragma unroll
        for (int j = 0; j < 4; ++j) {
            c0[j] = WMMA_BF16(a0.v, bf[j].v, c0[j]);
            c1[j] = WMMA_BF16(a1.v, bf[j].v, c1[j]);
        }
        a0 = a0n; a1 = a1n;
    }
    int half = hi ? 8 : 0;
#pragma unroll
    for (int sub = 0; sub < 2; ++sub) {
#pragma unroll
        for (int j = 0; j < 4; ++j) {
            int o = ng * 64 + j * 16 + ln;         // output feature 0..2303
            int sect = o / Cn;                     // 0=Q 1=K 2=V
            int hh = (o % Cn) / Dn;
            int dk = o % Dn;
#pragma unroll
            for (int r = 0; r < 8; ++r) {
                int grow = mt2 * 32 + sub * 16 + half + r;   // global row 0..8191
                int bb = grow >> 10, l = grow & (Ln - 1);
                float cv = sub ? c1[j][r] : c0[j][r];
                __bf16 val = f2bf(cv);
                if (sect == 0)      Qg[(((size_t)(bb * Hn + hh)) * Ln + l) * Dn + dk] = val;
                else if (sect == 1) Kg[(((size_t)(bb * Hn + hh)) * Ln + l) * Dn + dk] = val;
                else                Vt[(((size_t)(bb * Hn + hh)) * Dn + dk) * Ln + l] = val;
            }
        }
    }
}

// ---------------- flash attention: one wave per (b, h, 16-row q tile) ----------------
__global__ void __launch_bounds__(128, 1)
attn_kernel(const unsigned short* __restrict__ Qg_u,
            const unsigned short* __restrict__ Kg_u,
            const unsigned short* __restrict__ Vt_u,
            const unsigned char* __restrict__ maskb,
            unsigned short* __restrict__ Ob_u) {
    const __bf16* Qg = (const __bf16*)Qg_u;
    const __bf16* Kg = (const __bf16*)Kg_u;
    const __bf16* Vt = (const __bf16*)Vt_u;
    __bf16* Ob = (__bf16*)Ob_u;

    __shared__ __bf16 plds[4][16 * 32];
    int lane = threadIdx.x & 31;
    int widx = threadIdx.x >> 5;
    int wg = blockIdx.x * 4 + widx;          // B*H*64 waves
    int qt = wg & 63;
    int h  = (wg >> 6) % Hn;
    int b  = wg / (64 * Hn);
    int ln = lane & 15;
    bool hi = lane >= 16;
    int half = hi ? 8 : 0;

    const __bf16* qbase = Qg + ((size_t)(b * Hn + h)) * Ln * Dn;
    const __bf16* kbase = Kg + ((size_t)(b * Hn + h)) * Ln * Dn;
    const __bf16* vbase = Vt + ((size_t)(b * Hn + h)) * Dn * Ln;

    // Q fragments (A-operand), 4 chunks of K=32
    V16 qa[4];
    {
        const __bf16* qp = qbase + (size_t)(qt * 16 + ln) * Dn + (hi ? 8 : 0);
#pragma unroll
        for (int cc = 0; cc < 4; ++cc) {
            qa[cc].h[0] = *(const v8bf*)(qp + cc * 32);
            qa[cc].h[1] = *(const v8bf*)(qp + cc * 32 + 16);
        }
    }
    const unsigned char* mptr[8];
#pragma unroll
    for (int r = 0; r < 8; ++r)
        mptr[r] = maskb + ((size_t)b * Ln + qt * 16 + half + r) * Ln;

    v8f acc[8];
#pragma unroll
    for (int j = 0; j < 8; ++j)
#pragma unroll
        for (int e = 0; e < 8; ++e) acc[j][e] = 0.f;
    float mrun[8], lrun[8];
#pragma unroll
    for (int r = 0; r < 8; ++r) { mrun[r] = -3e38f; lrun[r] = 0.f; }

    const float scale = 0.08838834764831845f;   // 1/sqrt(128)

    for (int kb = 0; kb < Ln; kb += 32) {
        // prefetch next key tile: one 256B key row per lane (2 cachelines)
        if (kb + 32 < Ln) {
            const __bf16* pk = kbase + (size_t)(kb + 32 + lane) * Dn;
            __builtin_prefetch(pk, 0, 1);
            __builtin_prefetch(pk + 64, 0, 1);
        }
        // S = Q @ K^T for 32 keys (two 16-key subtiles)
        v8f s0, s1;
#pragma unroll
        for (int e = 0; e < 8; ++e) { s0[e] = 0.f; s1[e] = 0.f; }
        {
            const __bf16* kp0 = kbase + (size_t)(kb + ln) * Dn + (hi ? 8 : 0);
            const __bf16* kp1 = kbase + (size_t)(kb + 16 + ln) * Dn + (hi ? 8 : 0);
#pragma unroll
            for (int cc = 0; cc < 4; ++cc) {
                V16 bf0, bf1;
                bf0.h[0] = *(const v8bf*)(kp0 + cc * 32);
                bf0.h[1] = *(const v8bf*)(kp0 + cc * 32 + 16);
                bf1.h[0] = *(const v8bf*)(kp1 + cc * 32);
                bf1.h[1] = *(const v8bf*)(kp1 + cc * 32 + 16);
                s0 = WMMA_BF16(qa[cc].v, bf0.v, s0);
                s1 = WMMA_BF16(qa[cc].v, bf1.v, s1);
            }
        }
        // scale + structural mask + row max
        float p0[8], p1[8], tmax[8];
#pragma unroll
        for (int r = 0; r < 8; ++r) {
            float x0 = s0[r] * scale, x1 = s1[r] * scale;
            unsigned m0 = mptr[r][kb + ln];
            unsigned m1 = mptr[r][kb + 16 + ln];
            if (!((m0 >> h) & 1u)) x0 = -1e9f;
            if (!((m1 >> h) & 1u)) x1 = -1e9f;
            p0[r] = x0; p1[r] = x1;
            tmax[r] = fmaxf(x0, x1);
        }
#pragma unroll
        for (int dd = 1; dd < 16; dd <<= 1)
#pragma unroll
            for (int r = 0; r < 8; ++r)
                tmax[r] = fmaxf(tmax[r], __shfl_xor(tmax[r], dd, 32));
        // online softmax update
        float alpha[8], rs[8];
#pragma unroll
        for (int r = 0; r < 8; ++r) {
            float mn = fmaxf(mrun[r], tmax[r]);
            alpha[r] = __expf(mrun[r] - mn);
            mrun[r] = mn;
            p0[r] = __expf(p0[r] - mn);
            p1[r] = __expf(p1[r] - mn);
            rs[r] = p0[r] + p1[r];
        }
#pragma unroll
        for (int dd = 1; dd < 16; dd <<= 1)
#pragma unroll
            for (int r = 0; r < 8; ++r)
                rs[r] += __shfl_xor(rs[r], dd, 32);
#pragma unroll
        for (int r = 0; r < 8; ++r) lrun[r] = lrun[r] * alpha[r] + rs[r];
#pragma unroll
        for (int j = 0; j < 8; ++j)
#pragma unroll
            for (int r = 0; r < 8; ++r) acc[j][r] *= alpha[r];

        // P (16x32) -> LDS -> A-operand layout
        __bf16* pl = plds[widx];
#pragma unroll
        for (int r = 0; r < 8; ++r) {
            pl[(half + r) * 32 + ln]      = f2bf(p0[r]);
            pl[(half + r) * 32 + 16 + ln] = f2bf(p1[r]);
        }
        asm volatile("s_wait_dscnt 0" ::: "memory");
        V16 pa;
        {
            const __bf16* pp = pl + ln * 32 + (hi ? 8 : 0);
            pa.h[0] = *(const v8bf*)(pp);
            pa.h[1] = *(const v8bf*)(pp + 16);
        }
        // O += P @ V  (V stored transposed: Vt[feature][key])
        const __bf16* vp = vbase + kb + (hi ? 8 : 0);
#pragma unroll
        for (int j = 0; j < 8; ++j) {
            const __bf16* vr = vp + (size_t)(j * 16 + ln) * Ln;
            V16 bfr;
            bfr.h[0] = *(const v8bf*)(vr);
            bfr.h[1] = *(const v8bf*)(vr + 16);
            acc[j] = WMMA_BF16(pa.v, bfr.v, acc[j]);
        }
    }
    // normalize + store to (b,l,768) bf16 for the projection GEMM
#pragma unroll
    for (int r = 0; r < 8; ++r) {
        float inv = 1.0f / lrun[r];
        int grow = qt * 16 + half + r;
        __bf16* op = Ob + ((size_t)(b * Ln + grow)) * Cn + h * Dn + ln;
#pragma unroll
        for (int j = 0; j < 8; ++j)
            op[j * 16] = f2bf(acc[j][r] * inv);
    }
}

// ---------------- output projection: 32x64 tile per wave -> fp32 out ----------------
__global__ void __launch_bounds__(128, 1)
gemm_proj(const unsigned short* __restrict__ Ob_u,
          const unsigned short* __restrict__ wb_u,
          float* __restrict__ out) {
    const __bf16* Ob = (const __bf16*)Ob_u;
    const __bf16* wb = (const __bf16*)wb_u;
    int lane = threadIdx.x & 31;
    int wg = blockIdx.x * 4 + (threadIdx.x >> 5);   // 256*12 waves
    int mt2 = wg & 255;
    int ng = wg >> 8;                               // 0..11
    int ln = lane & 15;
    bool hi = lane >= 16;

    v8f c0[4], c1[4];
#pragma unroll
    for (int j = 0; j < 4; ++j)
#pragma unroll
        for (int e = 0; e < 8; ++e) { c0[j][e] = 0.f; c1[j][e] = 0.f; }

    const __bf16* ap0 = Ob + (size_t)(mt2 * 32 + ln) * Cn + (hi ? 8 : 0);
    const __bf16* ap1 = ap0 + (size_t)16 * Cn;
    const __bf16* bb0 = wb + (size_t)(ng * 64 + ln) * Cn + (hi ? 8 : 0);

    V16 a0 = loadA(ap0, 0), a1 = loadA(ap1, 0);
#pragma unroll 2
    for (int kc = 0; kc < Cn / 32; ++kc) {
        V16 bf[4];
#pragma unroll
        for (int j = 0; j < 4; ++j)
            bf[j] = loadB(bb0 + (size_t)(j * 16) * Cn + kc * 32);
        int kn = (kc == Cn / 32 - 1) ? 0 : kc + 1;
        V16 a0n = loadA(ap0, kn), a1n = loadA(ap1, kn);
#pragma unroll
        for (int j = 0; j < 4; ++j) {
            c0[j] = WMMA_BF16(a0.v, bf[j].v, c0[j]);
            c1[j] = WMMA_BF16(a1.v, bf[j].v, c1[j]);
        }
        a0 = a0n; a1 = a1n;
    }
    int half = hi ? 8 : 0;
#pragma unroll
    for (int sub = 0; sub < 2; ++sub) {
#pragma unroll
        for (int j = 0; j < 4; ++j) {
            int n = ng * 64 + j * 16 + ln;
#pragma unroll
            for (int r = 0; r < 8; ++r) {
                int grow = mt2 * 32 + sub * 16 + half + r;
                out[(size_t)grow * Cn + n] = sub ? c1[j][r] : c0[j][r];
            }
        }
    }
}

// ---------------- launch ----------------
extern "C" void kernel_launch(void* const* d_in, const int* in_sizes, int n_in,
                              void* d_out, int out_size, void* d_ws, size_t ws_size,
                              hipStream_t stream) {
    (void)in_sizes; (void)n_in; (void)out_size; (void)ws_size;
    const float* x        = (const float*)d_in[0];   // (8,1024,768)
    const int*   adj      = (const int*)d_in[1];     // (8,1024,1024)
    const int*   distance = (const int*)d_in[2];     // (8,1024,1024)
    const float* w_qkv    = (const float*)d_in[3];   // (2304,768)
    const float* w_proj   = (const float*)d_in[4];   // (768,768)
    float* out = (float*)d_out;                      // (8,1024,768)

    char* ws = (char*)d_ws;
    unsigned short* xb    = (unsigned short*)ws; ws += (size_t)Bn * Ln * Cn * 2;         // 12.6 MB
    unsigned short* wqkvb = (unsigned short*)ws; ws += (size_t)3 * Cn * Cn * 2;          // 3.5 MB
    unsigned short* wprjb = (unsigned short*)ws; ws += (size_t)Cn * Cn * 2;              // 1.1 MB
    unsigned int*   a1    = (unsigned int*)ws;   ws += (size_t)Bn * Ln * 32 * 4;         // 1 MB
    unsigned int*   a2    = (unsigned int*)ws;   ws += (size_t)Bn * Ln * 32 * 4;         // 1 MB
    unsigned char*  maskb = (unsigned char*)ws;  ws += (size_t)Bn * Ln * Ln;             // 8 MB
    unsigned short* Qg    = (unsigned short*)ws; ws += (size_t)Bn * Hn * Ln * Dn * 2;    // 12.6 MB
    unsigned short* Kg    = (unsigned short*)ws; ws += (size_t)Bn * Hn * Ln * Dn * 2;
    unsigned short* Vt    = (unsigned short*)ws; ws += (size_t)Bn * Hn * Ln * Dn * 2;
    unsigned short* Ob    = (unsigned short*)ws; ws += (size_t)Bn * Ln * Cn * 2;

    int nx = Bn * Ln * Cn;            // 6291456
    int nwq = 3 * Cn * Cn;            // 1769472
    int nwp = Cn * Cn;                // 589824
    cvt_bf16<<<(nx + 255) / 256, 256, 0, stream>>>(x, xb, nx);
    cvt_bf16<<<(nwq + 255) / 256, 256, 0, stream>>>(w_qkv, wqkvb, nwq);
    cvt_bf16<<<(nwp + 255) / 256, 256, 0, stream>>>(w_proj, wprjb, nwp);

    build_a1<<<(Bn * Ln * 32) / 256, 256, 0, stream>>>(adj, a1);
    build_a2<<<(Bn * Ln) / 4, 128, 0, stream>>>(a1, a2);
    build_mask<<<(Bn * Ln * Ln) / 256, 256, 0, stream>>>(distance, a1, a2, maskb);

    gemm_qkv<<<(256 * 36) / 4, 128, 0, stream>>>(xb, wqkvb, Qg, Kg, Vt);
    attn_kernel<<<(Bn * Hn * 64) / 4, 128, 0, stream>>>(Qg, Kg, Vt, maskb, Ob);
    gemm_proj<<<(256 * 12) / 4, 128, 0, stream>>>(Ob, wprjb, out);
}
